// TFXLNetMainLayer_24644522344543
// MI455X (gfx1250) — compile-verified, matrix operand order checked by reading, not run on hardware
//
#include <hip/hip_runtime.h>
#include <cmath>

typedef __attribute__((ext_vector_type(16))) _Float16 v16h;
typedef __attribute__((ext_vector_type(8)))  _Float16 v8h;
typedef __attribute__((ext_vector_type(8)))  float    v8f;

// ---------------- problem constants ----------------
constexpr int Qn  = 2048;   // qlen = klen
constexpr int Dm  = 1024;   // d_model
constexpr int NH  = 16;     // heads
constexpr int DHd = 64;     // d_head
constexpr int DIn = 4096;   // d_inner
constexpr int Rn  = 4096;   // rlen = 2*Q
constexpr int LAY = 2;
constexpr float ATT_SCALE = 0.125f;   // 1/sqrt(64)

// ---------------- GEMM config ----------------
// Block tile 256x64, 8 waves stacked in M, each wave 32x64 (2x4 WMMA tiles).
constexpr int BM = 256, BN = 64, BK = 32, LDK = 40;   // LDS K-stride padded (80B rows)
constexpr int EF32 = 1, EF16 = 2, EBIAS = 4, EGELU = 8, ERESID = 16;

// Load one 16x32 f16 WMMA fragment from an LDS tile stored [row][LDK] halves.
// CDNA5 16-bit A/B layout: lane<16 holds K{0..7,16..23}, lane>=16 holds K{8..15,24..31};
// vector elements 0..7 are one contiguous K-oct, elements 8..15 the oct at K+16.
__device__ __forceinline__ v16h ld_frag(const _Float16* s, int rowBase, int lane) {
  const int row  = rowBase + (lane & 15);
  const int koff = (lane & 16) ? 8 : 0;
  const _Float16* p = s + row * LDK + koff;
  v8h lo = *(const v8h*)(p);
  v8h hi = *(const v8h*)(p + 16);
  v16h r;
#pragma unroll
  for (int e = 0; e < 8; ++e) { r[e] = lo[e]; r[e + 8] = hi[e]; }
  return r;
}

__device__ __forceinline__ float wave_red_sum(float v) {
#pragma unroll
  for (int o = 16; o > 0; o >>= 1) v += __shfl_xor(v, o, 32);
  return v;
}
__device__ __forceinline__ float wave_red_max(float v) {
#pragma unroll
  for (int o = 16; o > 0; o >>= 1) v = fmaxf(v, __shfl_xor(v, o, 32));
  return v;
}

// C[M,N](f32 and/or f16) = A[M,K](f16,row-major) * B[N,K](f16,row-major)^T  (NT GEMM)
// Double-buffered LDS tiles, one barrier per K-step, prefetch of K+2 tile.
// Epilogue flags: +bias[col], exact GELU, +resid[row*ldc+col].
template <int EPI>
__global__ __launch_bounds__(256) void gemm_nt(const _Float16* __restrict__ A, int lda,
                                               const _Float16* __restrict__ B, int ldb,
                                               float* __restrict__ C, _Float16* __restrict__ C16,
                                               int ldc, const float* __restrict__ bias,
                                               const float* __restrict__ resid, int K) {
  __shared__ __align__(16) _Float16 sA[2][BM * LDK];
  __shared__ __align__(16) _Float16 sB[2][BN * LDK];
  const int tid  = threadIdx.x;
  const int lane = tid & 31;
  const int wave = tid >> 5;                        // 0..7 -> 32-row strip
  const long rowBase = (long)blockIdx.y * BM;
  const long colBase = (long)blockIdx.x * BN;

  v8f acc[2][4] = {};

  // global -> LDS staging: A = 1024 v8h chunks (4/thread), B = 256 chunks (1/thread)
  auto load_tile = [&](int k0, _Float16* bufA, _Float16* bufB) {
#pragma unroll
    for (int it = 0; it < (BM * BK / 8) / 256; ++it) {           // 4 iterations
      const int c = tid + it * 256;
      const int r = c >> 2, kc = (c & 3) << 3;
      *(v8h*)&bufA[r * LDK + kc] = *(const v8h*)&A[(rowBase + r) * lda + k0 + kc];
    }
    {
      const int r = tid >> 2, kc = (tid & 3) << 3;               // 64x32 B tile
      *(v8h*)&bufB[r * LDK + kc] = *(const v8h*)&B[(colBase + r) * ldb + k0 + kc];
    }
  };

  auto compute = [&](const _Float16* bufA, const _Float16* bufB) {
    const v16h a0 = ld_frag(bufA, wave * 32, lane);
    const v16h a1 = ld_frag(bufA, wave * 32 + 16, lane);
    v16h b[4];
#pragma unroll
    for (int t = 0; t < 4; ++t) b[t] = ld_frag(bufB, t * 16, lane);
#pragma unroll
    for (int t = 0; t < 4; ++t) {
      acc[0][t] = __builtin_amdgcn_wmma_f32_16x16x32_f16(false, a0, false, b[t], (short)0, acc[0][t], false, false);
      acc[1][t] = __builtin_amdgcn_wmma_f32_16x16x32_f16(false, a1, false, b[t], (short)0, acc[1][t], false, false);
    }
  };

  const int nk = K / BK;
  load_tile(0, sA[0], sB[0]);
  __syncthreads();
  for (int kt = 0; kt < nk; ++kt) {
    const int cur = kt & 1;
    if (kt + 1 < nk) {
      load_tile((kt + 1) * BK, sA[cur ^ 1], sB[cur ^ 1]);
      if (kt + 2 < nk) {                                        // pull K+2 lines toward cache
        const int kp = (kt + 2) * BK;
        __builtin_prefetch(&A[(rowBase + tid) * lda + kp], 0, 0);
        if (tid < BN) __builtin_prefetch(&B[(colBase + tid) * ldb + kp], 0, 0);
      }
    }
    compute(sA[cur], sB[cur]);
    __syncthreads();
  }

  // C/D layout: lane&15 = column, VGPR v = row v (+8 for lane>=16)
  const int lcol = lane & 15;
  const int radd = (lane >> 4) << 3;
#pragma unroll
  for (int ti = 0; ti < 2; ++ti)
#pragma unroll
    for (int tj = 0; tj < 4; ++tj) {
      const long gcol = colBase + tj * 16 + lcol;
#pragma unroll
      for (int v = 0; v < 8; ++v) {
        const long grow = rowBase + wave * 32 + ti * 16 + radd + v;
        float x = acc[ti][tj][v];
        if constexpr (EPI & EBIAS)  x += bias[gcol];
        if constexpr (EPI & EGELU)  x = 0.5f * x * (1.0f + erff(x * 0.70710678118654752f));
        if constexpr (EPI & ERESID) x += resid[grow * ldc + gcol];
        if constexpr (EPI & EF32)   C[grow * ldc + gcol] = x;
        if constexpr (EPI & EF16)   C16[grow * ldc + gcol] = (_Float16)x;
      }
    }
}

// dst[c][r] (f16) = src[r][c] (f32); z-batched with element offsets srcZ/dstZ.
__global__ void transpose_cast_kernel(const float* __restrict__ src, _Float16* __restrict__ dst,
                                      int srcLd, int dstLd, long srcZ, long dstZ) {
  __shared__ float t[32][33];
  const long z = blockIdx.z;
  const float* s = src + z * srcZ;
  _Float16*    d = dst + z * dstZ;
  const int c0 = blockIdx.x * 32, r0 = blockIdx.y * 32;
#pragma unroll
  for (int it = 0; it < 4; ++it) {
    const int rr = threadIdx.y + it * 8;
    t[rr][threadIdx.x] = s[(long)(r0 + rr) * srcLd + c0 + threadIdx.x];
  }
  __syncthreads();
#pragma unroll
  for (int it = 0; it < 4; ++it) {
    const int rr = threadIdx.y + it * 8;
    d[(long)(c0 + rr) * dstLd + r0 + threadIdx.x] = (_Float16)t[threadIdx.x][rr];
  }
}

__global__ void cast_f32_f16_kernel(const float* __restrict__ src, _Float16* __restrict__ dst, long n) {
  long i = (long)blockIdx.x * blockDim.x + threadIdx.x;
  const long stride = (long)gridDim.x * blockDim.x;
  for (; i < n; i += stride) dst[i] = (_Float16)src[i];
}

// From q_head(f32) build f16 (q+r_w_bias), (q+r_r_bias), and ef[i,n,s] = sum_d (q+r_s_bias)*seg_embed[s].
__global__ void qprep_kernel(const float* __restrict__ qh, const float* __restrict__ rwb,
                             const float* __restrict__ rrb, const float* __restrict__ rsb,
                             const float* __restrict__ seg_embed,
                             _Float16* __restrict__ qw16, _Float16* __restrict__ qr16,
                             float* __restrict__ ef) {
  const int idx = blockIdx.x * blockDim.x + threadIdx.x;  // i*NH + n
  const int i = idx >> 4, n = idx & 15;
  const int base = i * Dm + n * DHd, bb = n * DHd;
  float e0 = 0.f, e1 = 0.f;
#pragma unroll 8
  for (int d = 0; d < DHd; ++d) {
    const float q = qh[base + d];
    qw16[base + d] = (_Float16)(q + rwb[bb + d]);
    qr16[base + d] = (_Float16)(q + rrb[bb + d]);
    const float qs = q + rsb[bb + d];
    e0 += qs * seg_embed[bb + d];           // seg_embed[0][n][d]
    e1 += qs * seg_embed[Dm + bb + d];      // seg_embed[1][n][d]  (NH*DH == Dm)
  }
  ef[(long)idx * 2 + 0] = e0;
  ef[(long)idx * 2 + 1] = e1;
}

// Row-wise softmax of (ac + rel_shift(bd) + ef_gather)*SCALE with causal mask (j<=i). Writes f16 P.
__global__ __launch_bounds__(256) void softmax_kernel(const float* __restrict__ ac,
                                                      const float* __restrict__ bdraw,
                                                      const float* __restrict__ ef,
                                                      const int* __restrict__ seg,
                                                      _Float16* __restrict__ p16, int head) {
  __shared__ float red[8];
  const int i = blockIdx.x, tid = threadIdx.x;
  const int lane = tid & 31, wave = tid >> 5;
  const int nv = i + 1;                          // valid keys (causal)
  const int si = seg[i];
  const float e0 = ef[((long)i * NH + head) * 2 + 0];
  const float e1 = ef[((long)i * NH + head) * 2 + 1];
  float vals[8];
  float lmax = -3.0e38f;
#pragma unroll
  for (int t = 0; t < 8; ++t) {
    const int j = tid + t * 256;
    float s = -3.0e38f;
    if (j < nv) {
      // rel_shift: bd[i,j] = raw_bd[i, j + Q - i]
      s = ac[(long)i * Qn + j] + bdraw[(long)i * Rn + (j + Qn - i)];
      s += (seg[j] != si) ? e1 : e0;
      s *= ATT_SCALE;
    }
    vals[t] = s;
    lmax = fmaxf(lmax, s);
  }
  lmax = wave_red_max(lmax);
  if (lane == 0) red[wave] = lmax;
  __syncthreads();
  float m = red[0];
#pragma unroll
  for (int w = 1; w < 8; ++w) m = fmaxf(m, red[w]);
  __syncthreads();                                // red reused below
  float lsum = 0.f;
#pragma unroll
  for (int t = 0; t < 8; ++t) {
    const int j = tid + t * 256;
    float e = 0.f;
    if (j < nv) e = __expf(vals[t] - m);
    vals[t] = e; lsum += e;
  }
  lsum = wave_red_sum(lsum);
  if (lane == 0) red[wave] = lsum;
  __syncthreads();
  float total = 0.f;
#pragma unroll
  for (int w = 0; w < 8; ++w) total += red[w];
  const float inv = 1.0f / total;
#pragma unroll
  for (int t = 0; t < 8; ++t) {
    const int j = tid + t * 256;
    p16[(long)i * Qn + j] = (_Float16)(vals[t] * inv);   // j>=nv writes exact 0
  }
}

// y = LN(x)*g + b, writes both f32 state and f16 copy for next GEMM. One block per row, D=1024.
__global__ __launch_bounds__(256) void layernorm_kernel(const float* __restrict__ x,
                                                        const float* __restrict__ g,
                                                        const float* __restrict__ b,
                                                        float* __restrict__ y,
                                                        _Float16* __restrict__ y16) {
  __shared__ float red[8];
  const int row = blockIdx.x, tid = threadIdx.x;
  const int lane = tid & 31, wave = tid >> 5;
  const float* xr = x + (long)row * Dm;
  float v[4]; float s = 0.f;
#pragma unroll
  for (int t = 0; t < 4; ++t) { v[t] = xr[tid + t * 256]; s += v[t]; }
  s = wave_red_sum(s);
  if (lane == 0) red[wave] = s;
  __syncthreads();
  float tot = 0.f;
#pragma unroll
  for (int w = 0; w < 8; ++w) tot += red[w];
  const float mean = tot * (1.0f / Dm);
  __syncthreads();                                // red reused below
  float vs = 0.f;
#pragma unroll
  for (int t = 0; t < 4; ++t) { const float d = v[t] - mean; vs += d * d; }
  vs = wave_red_sum(vs);
  if (lane == 0) red[wave] = vs;
  __syncthreads();
  float vtot = 0.f;
#pragma unroll
  for (int w = 0; w < 8; ++w) vtot += red[w];
  const float inv = rsqrtf(vtot * (1.0f / Dm) + 1e-12f);
#pragma unroll
  for (int t = 0; t < 4; ++t) {
    const int col = tid + t * 256;
    const float yv = (v[t] - mean) * inv * g[col] + b[col];
    y[(long)row * Dm + col] = yv;
    y16[(long)row * Dm + col] = (_Float16)yv;
  }
}

template <int EPI>
static void launch_gemm(hipStream_t s, const _Float16* A, int lda, const _Float16* B, int ldb,
                        float* C, _Float16* C16, int ldc, const float* bias, const float* resid,
                        int M, int N, int K) {
  dim3 grid(N / BN, M / BM), blk(256);
  gemm_nt<EPI><<<grid, blk, 0, s>>>(A, lda, B, ldb, C, C16, ldc, bias, resid, K);
}

extern "C" void kernel_launch(void* const* d_in, const int* in_sizes, int n_in,
                              void* d_out, int out_size, void* d_ws, size_t ws_size,
                              hipStream_t stream) {
  (void)in_sizes; (void)n_in; (void)ws_size;
  const float* in_h      = (const float*)d_in[0];
  const float* pos_emb   = (const float*)d_in[1];
  const int*   seg_ids   = (const int*)d_in[2];
  // d_in[3] attn_mask: exactly the causal mask (j>i); applied analytically in softmax.
  const float* Wq        = (const float*)d_in[4];
  const float* Wk        = (const float*)d_in[5];
  const float* Wv        = (const float*)d_in[6];
  const float* Wo        = (const float*)d_in[7];
  const float* Wr        = (const float*)d_in[8];
  const float* rwb       = (const float*)d_in[9];
  const float* rrb       = (const float*)d_in[10];
  const float* rsb       = (const float*)d_in[11];
  const float* seg_embed = (const float*)d_in[12];
  const float* aln_g     = (const float*)d_in[13];
  const float* aln_b     = (const float*)d_in[14];
  const float* ffw1      = (const float*)d_in[15];
  const float* ffb1      = (const float*)d_in[16];
  const float* ffw2      = (const float*)d_in[17];
  const float* ffb2      = (const float*)d_in[18];
  const float* fln_g     = (const float*)d_in[19];
  const float* fln_b     = (const float*)d_in[20];

  char* wsb = (char*)d_ws;
  size_t off = 0;
  auto alloc = [&](size_t bytes) -> void* {
    void* p = wsb + off;
    off = (off + bytes + 255) & ~(size_t)255;
    return p;
  };
  float*    h     = (float*)alloc((size_t)Qn * Dm * 4);
  _Float16* h16   = (_Float16*)alloc((size_t)Qn * Dm * 2);
  _Float16* pos16 = (_Float16*)alloc((size_t)Rn * Dm * 2);
  _Float16* wqT   = (_Float16*)alloc((size_t)Dm * Dm * 2);
  _Float16* wkT   = (_Float16*)alloc((size_t)Dm * Dm * 2);
  _Float16* wvT   = (_Float16*)alloc((size_t)Dm * Dm * 2);
  _Float16* wrT   = (_Float16*)alloc((size_t)Dm * Dm * 2);
  _Float16* wo16  = (_Float16*)alloc((size_t)Dm * Dm * 2);
  _Float16* w1T   = (_Float16*)alloc((size_t)DIn * Dm * 2);
  _Float16* w2T   = (_Float16*)alloc((size_t)Dm * DIn * 2);
  float*    qh    = (float*)alloc((size_t)Qn * Dm * 4);
  _Float16* kh16  = (_Float16*)alloc((size_t)Qn * Dm * 2);
  float*    vh    = (float*)alloc((size_t)Qn * Dm * 4);
  _Float16* vT16  = (_Float16*)alloc((size_t)Dm * Qn * 2);   // [NH][DH][Q]
  _Float16* krh16 = (_Float16*)alloc((size_t)Rn * Dm * 2);
  _Float16* qw16  = (_Float16*)alloc((size_t)Qn * Dm * 2);
  _Float16* qr16  = (_Float16*)alloc((size_t)Qn * Dm * 2);
  float*    ef    = (float*)alloc((size_t)Qn * NH * 2 * 4);
  float*    acb   = (float*)alloc((size_t)Qn * Qn * 4);
  float*    bdraw = (float*)alloc((size_t)Qn * Rn * 4);
  _Float16* p16   = (_Float16*)alloc((size_t)Qn * Qn * 2);
  _Float16* av16  = (_Float16*)alloc((size_t)Qn * Dm * 2);
  float*    preln = (float*)alloc((size_t)Qn * Dm * 4);
  _Float16* mid16 = (_Float16*)alloc((size_t)Qn * DIn * 2);

  auto cast16 = [&](const float* src, _Float16* dst, long n) {
    const int blocks = (int)((n + 1023) / 1024);
    cast_f32_f16_kernel<<<blocks, 256, 0, stream>>>(src, dst, n);
  };
  auto tposeN = [&](const float* src, _Float16* dst, int rows, int cols, int srcLd, int dstLd,
                    int nz, long srcZ, long dstZ) {
    dim3 g(cols / 32, rows / 32, nz), b(32, 8);
    transpose_cast_kernel<<<g, b, 0, stream>>>(src, dst, srcLd, dstLd, srcZ, dstZ);
  };

  hipMemcpyAsync(h, in_h, (size_t)Qn * Dm * 4, hipMemcpyDeviceToDevice, stream);
  cast16(in_h, h16, (long)Qn * Dm);
  cast16(pos_emb, pos16, (long)Rn * Dm);

  for (int l = 0; l < LAY; ++l) {
    const float* Wq_l = Wq + (size_t)l * Dm * Dm;
    const float* Wk_l = Wk + (size_t)l * Dm * Dm;
    const float* Wv_l = Wv + (size_t)l * Dm * Dm;
    const float* Wo_l = Wo + (size_t)l * Dm * Dm;
    const float* Wr_l = Wr + (size_t)l * Dm * Dm;
    const float* rwb_l = rwb + (size_t)l * NH * DHd;
    const float* rrb_l = rrb + (size_t)l * NH * DHd;
    const float* rsb_l = rsb + (size_t)l * NH * DHd;
    const float* se_l  = seg_embed + (size_t)l * 2 * NH * DHd;
    const float* w1_l  = ffw1 + (size_t)l * Dm * DIn;
    const float* b1_l  = ffb1 + (size_t)l * DIn;
    const float* w2_l  = ffw2 + (size_t)l * DIn * Dm;
    const float* b2_l  = ffb2 + (size_t)l * Dm;

    // --- weight prep: transposed [out][in] f16 copies (B operand of NT GEMM) ---
    tposeN(Wq_l, wqT, Dm, Dm, Dm, Dm, 1, 0, 0);
    tposeN(Wk_l, wkT, Dm, Dm, Dm, Dm, 1, 0, 0);
    tposeN(Wv_l, wvT, Dm, Dm, Dm, Dm, 1, 0, 0);
    tposeN(Wr_l, wrT, Dm, Dm, Dm, Dm, 1, 0, 0);
    cast16(Wo_l, wo16, (long)Dm * Dm);                    // Wo[h][n*d] is already B-layout
    tposeN(w1_l, w1T, Dm, DIn, DIn, Dm, 1, 0, 0);
    tposeN(w2_l, w2T, DIn, Dm, Dm, DIn, 1, 0, 0);

    // --- projections (all WMMA) ---
    launch_gemm<EF32>(stream, h16, Dm, wqT, Dm, qh, nullptr, Dm, nullptr, nullptr, Qn, Dm, Dm);
    launch_gemm<EF16>(stream, h16, Dm, wkT, Dm, nullptr, kh16, Dm, nullptr, nullptr, Qn, Dm, Dm);
    launch_gemm<EF32>(stream, h16, Dm, wvT, Dm, vh, nullptr, Dm, nullptr, nullptr, Qn, Dm, Dm);
    launch_gemm<EF16>(stream, pos16, Dm, wrT, Dm, nullptr, krh16, Dm, nullptr, nullptr, Rn, Dm, Dm);

    // per-head V^T : [Q,64] -> [64,Q], z = head
    tposeN(vh, vT16, Qn, DHd, Dm, Qn, NH, DHd, (long)DHd * Qn);

    qprep_kernel<<<(Qn * NH) / 256, 256, 0, stream>>>(qh, rwb_l, rrb_l, rsb_l, se_l, qw16, qr16, ef);

    // --- attention, streamed per head (score buffers reused) ---
    for (int n = 0; n < NH; ++n) {
      launch_gemm<EF32>(stream, qw16 + n * DHd, Dm, kh16 + n * DHd, Dm,
                        acb, nullptr, Qn, nullptr, nullptr, Qn, Qn, DHd);
      launch_gemm<EF32>(stream, qr16 + n * DHd, Dm, krh16 + n * DHd, Dm,
                        bdraw, nullptr, Rn, nullptr, nullptr, Qn, Rn, DHd);
      softmax_kernel<<<Qn, 256, 0, stream>>>(acb, bdraw, ef, seg_ids, p16, n);
      launch_gemm<EF16>(stream, p16, Qn, vT16 + (size_t)n * DHd * Qn, Qn,
                        nullptr, av16 + n * DHd, Dm, nullptr, nullptr, Qn, DHd, Qn);
    }

    // --- output projection + residual + LN ---
    launch_gemm<EF32 | ERESID>(stream, av16, Dm, wo16, Dm, preln, nullptr, Dm, nullptr, h, Qn, Dm, Dm);
    layernorm_kernel<<<Qn, 256, 0, stream>>>(preln, aln_g + (size_t)l * Dm, aln_b + (size_t)l * Dm, h, h16);

    // --- FFN: gelu(h*W1 + b1)*W2 + b2 + h, then LN ---
    launch_gemm<EF16 | EBIAS | EGELU>(stream, h16, Dm, w1T, Dm, nullptr, mid16, DIn, b1_l, nullptr, Qn, DIn, Dm);
    launch_gemm<EF32 | EBIAS | ERESID>(stream, mid16, DIn, w2T, DIn, preln, nullptr, Dm, b2_l, h, Qn, Dm, DIn);
    layernorm_kernel<<<Qn, 256, 0, stream>>>(preln, fln_g + (size_t)l * Dm, fln_b + (size_t)l * Dm, h, h16);
  }

  hipMemcpyAsync(d_out, h, (size_t)out_size * sizeof(float), hipMemcpyDeviceToDevice, stream);
}
